// Tacotron2_34437047779864
// MI455X (gfx1250) — compile-verified, matrix operand order checked by reading, not run on hardware
//
#include <hip/hip_runtime.h>
#include <cmath>

typedef __attribute__((ext_vector_type(16))) _Float16 v16h;
typedef __attribute__((ext_vector_type(8)))  _Float16 v8h;
typedef __attribute__((ext_vector_type(8)))  float    v8f;

static __device__ __forceinline__ float sigf(float x){ return 1.f/(1.f+__expf(-x)); }

// ============================================================================
// WMMA GEMM:  C[M,N](f32,ldc) = act( A[M,K](f16,lda) * W[N,K](f16,ldw)^T
//                                    + bias[N] + (accum ? C : 0) )
// Optional secondary f16 output: F[row*ldf + fOff + col] = (f16)result.
// Shifted-row mode (conv-as-GEMM): source A row for output row m is m+tShift,
// valid iff 0 <= (m % Tdim) + tShift < Tdim (else zero fragment). Plain GEMM:
// tShift=0, Tdim=M.
// One wave computes a 16x32 C tile (1x2 N-tiles, shared A fragment -> two
// back-to-back independent v_wmma for XDL co-execution). wave32 only.
// K multiple of 32; M,N multiples of 16. Fragment layout per ISA 7.12.2.
// ============================================================================
__global__ __launch_bounds__(256) void k_wmma_gemm(
    const _Float16* __restrict__ A, int lda,
    const _Float16* __restrict__ W, int ldw,
    const float* __restrict__ bias,
    float* __restrict__ C, int ldc,
    int M, int N, int K, int tShift, int Tdim, int accum, int act,
    _Float16* __restrict__ F, int ldf, int fOff)
{
  int wave    = threadIdx.x >> 5;
  int lane    = threadIdx.x & 31;
  int tilesN  = N >> 4;
  int tilesN2 = (tilesN + 1) >> 1;           // N-tile pairs
  int tiles   = (M >> 4) * tilesN2;
  int tile    = blockIdx.x * (blockDim.x >> 5) + wave;
  if (tile >= tiles) return;                 // wave-uniform exit, EXEC stays full
  int tm    = (tile / tilesN2) << 4;
  int tn    = (tile % tilesN2) << 5;
  bool hasN2 = (tn + 16) < N;                // wave-uniform
  int half  = lane >> 4;                     // 0: lanes 0-15, 1: lanes 16-31
  int l16   = lane & 15;

  int m = tm + l16;                          // A row held by this lane
  bool valid = true;
  if (tShift != 0 || Tdim != M) {
    int t = m % Tdim;
    valid = (t + tShift >= 0) && (t + tShift < Tdim);
  }
  const _Float16* Arow  = A + (long long)(m + tShift) * lda;
  const _Float16* Wrow0 = W + (long long)(tn + l16) * ldw;
  const _Float16* Wrow1 = W + (long long)(tn + 16 + l16) * ldw;
  int ko = half * 8;                         // half0: K 0..7 & 16..23; half1: 8..15 & 24..31

  v8f acc0 = {};
  v8f acc1 = {};
  for (int kk = 0; kk < K; kk += 32) {
    v16h a = {};
    if (valid) {
      v8h a0 = *(const v8h*)(Arow + kk + ko);
      v8h a1 = *(const v8h*)(Arow + kk + 16 + ko);
      #pragma unroll
      for (int i = 0; i < 8; ++i) { a[i] = a0[i]; a[8+i] = a1[i]; }
    }
    v8h b0 = *(const v8h*)(Wrow0 + kk + ko);
    v8h b1 = *(const v8h*)(Wrow0 + kk + 16 + ko);
    v16h b;
    #pragma unroll
    for (int i = 0; i < 8; ++i) { b[i] = b0[i]; b[8+i] = b1[i]; }
    if (kk + 32 < K) {
      __builtin_prefetch((const void*)(Arow + kk + 32), 0, 0);  // global_prefetch_b8
      __builtin_prefetch((const void*)(Wrow0 + kk + 32), 0, 0);
    }
    acc0 = __builtin_amdgcn_wmma_f32_16x16x32_f16(false, a, false, b,
                                                  (short)0, acc0, false, false);
    if (hasN2) {                              // wave-uniform branch
      v8h c0 = *(const v8h*)(Wrow1 + kk + ko);
      v8h c1 = *(const v8h*)(Wrow1 + kk + 16 + ko);
      v16h b2;
      #pragma unroll
      for (int i = 0; i < 8; ++i) { b2[i] = c0[i]; b2[8+i] = c1[i]; }
      acc1 = __builtin_amdgcn_wmma_f32_16x16x32_f16(false, a, false, b2,
                                                    (short)0, acc1, false, false);
    }
  }

  #pragma unroll
  for (int r = 0; r < 8; ++r) {
    int row = tm + r + (half ? 8 : 0);       // C layout: VGPR r -> M=r (+8 hi half)
    {
      int col = tn + l16;
      long long o = (long long)row * ldc + col;
      float v = acc0[r];
      if (bias)  v += bias[col];
      if (accum) v += C[o];
      if (act == 1)      v = v > 0.f ? v : 0.f;
      else if (act == 2) v = tanhf(v);
      C[o] = v;
      if (F) F[(long long)row * ldf + fOff + col] = (_Float16)v;
    }
    if (hasN2) {
      int col = tn + 16 + l16;
      long long o = (long long)row * ldc + col;
      float v = acc1[r];
      if (bias)  v += bias[col];
      if (accum) v += C[o];
      if (act == 1)      v = v > 0.f ? v : 0.f;
      else if (act == 2) v = tanhf(v);
      C[o] = v;
      if (F) F[(long long)row * ldf + fOff + col] = (_Float16)v;
    }
  }
}

// ---------------- helpers / elementwise kernels ----------------
__global__ void k_fill32(float* p, float v, int n){
  int i = blockIdx.x*blockDim.x+threadIdx.x; if (i<n) p[i]=v;
}
__global__ void k_fill16(_Float16* p, int n){
  int i = blockIdx.x*blockDim.x+threadIdx.x; if (i<n) p[i]=(_Float16)0.f;
}
// f32 -> f16 with zero-padding to lddst columns
__global__ void k_cvt_pad(const float* __restrict__ s, int lds,
                          _Float16* __restrict__ d, int ldd, int rows, int cols){
  int i = blockIdx.x*blockDim.x+threadIdx.x; if (i >= rows*ldd) return;
  int r = i/ldd, c = i%ldd;
  d[i] = (_Float16)((c < cols) ? s[(long long)r*lds + c] : 0.f);
}
// slice conv weight [O,I,Kk] tap k -> f16 [O,ldd] zero padded
__global__ void k_convw_slice(const float* __restrict__ w, _Float16* __restrict__ d,
                              int O, int I, int Kk, int k, int ldd){
  int i = blockIdx.x*blockDim.x+threadIdx.x; if (i >= O*ldd) return;
  int o = i/ldd, c = i%ldd;
  d[i] = (_Float16)((c < I) ? w[((long long)o*I + c)*Kk + k] : 0.f);
}
// embedding gather -> f16 rows [B*T, E]
__global__ void k_embed(const int* __restrict__ text, const float* __restrict__ emb,
                        _Float16* __restrict__ x, int BT, int E){
  int i = blockIdx.x*blockDim.x+threadIdx.x; if (i >= BT*E) return;
  int r = i/E, c = i%E;
  x[i] = (_Float16)emb[(long long)text[r]*E + c];
}
// per-channel (column) mean / inv-std over R rows
__global__ void k_bn_stats(const float* __restrict__ x, int R, int C,
                           float* __restrict__ mu, float* __restrict__ inv, float eps){
  int c = blockIdx.x;
  __shared__ float s1[256], s2[256];
  float a = 0.f, b = 0.f;
  for (int r = threadIdx.x; r < R; r += blockDim.x){
    float v = x[(long long)r*C + c]; a += v; b += v*v;
  }
  s1[threadIdx.x]=a; s2[threadIdx.x]=b; __syncthreads();
  for (int s = blockDim.x>>1; s > 0; s >>= 1){
    if (threadIdx.x < s){ s1[threadIdx.x]+=s1[threadIdx.x+s]; s2[threadIdx.x]+=s2[threadIdx.x+s]; }
    __syncthreads();
  }
  if (threadIdx.x == 0){
    float m = s1[0]/R, v = s2[0]/R - m*m;
    mu[c] = m; inv[c] = rsqrtf(v + eps);
  }
}
// encoder BN apply: [B*T,C] (b-major) -> time-major f16 [T*B,C]
__global__ void k_enc_bn(const float* __restrict__ x, const float* mu, const float* inv,
                         const float* g, const float* be, _Float16* __restrict__ o,
                         int B, int T, int C){
  int i = blockIdx.x*blockDim.x+threadIdx.x; if (i >= B*T*C) return;
  int c = i%C, r = i/C, b = r/T, t = r%T;
  float v = (x[i]-mu[c])*inv[c]*g[c] + be[c];
  o[((long long)t*B + b)*C + c] = (_Float16)v;
}
// postnet BN + tanh -> f16 same layout
__global__ void k_bn_tanh(const float* __restrict__ x, const float* mu, const float* inv,
                          const float* g, const float* be, _Float16* __restrict__ o,
                          int R, int C){
  int i = blockIdx.x*blockDim.x+threadIdx.x; if (i >= R*C) return;
  int c = i%C;
  o[i] = (_Float16)tanhf((x[i]-mu[c])*inv[c]*g[c] + be[c]);
}
// LSTM cell pointwise. G row base already offset to step t. Writes next-step h
// (f16), layer output (time-major f16 [T*B,2H]), optionally memory f32/f16.
__global__ void k_lstm_point(const float* __restrict__ G, float* __restrict__ cb,
                             _Float16* __restrict__ hf16, _Float16* __restrict__ lout,
                             float* __restrict__ memF, _Float16* __restrict__ memH,
                             int t, int B, int H, int dOff, int T){
  int i = blockIdx.x*blockDim.x+threadIdx.x; if (i >= B*H) return;
  int b = i/H, j = i%H;
  const float* g = G + (long long)b*4*H;
  float cc = sigf(g[H+j])*cb[i] + sigf(g[j])*tanhf(g[2*H+j]);
  float hh = sigf(g[3*H+j])*tanhf(cc);
  cb[i] = cc;
  hf16[i] = (_Float16)hh;
  lout[((long long)t*B + b)*(2*H) + dOff + j] = (_Float16)hh;
  if (memF) memF[((long long)b*T + t)*(2*H) + dOff + j] = hh;
  if (memH) memH[((long long)b*T + t)*(2*H) + dOff + j] = (_Float16)hh;
}
// decoder LSTM cell: D=1024; writes new hd into concat1536[:,0:1024] (f16)
__global__ void k_dec_point(const float* __restrict__ g, float* __restrict__ cd,
                            _Float16* __restrict__ c1536, int B, int D){
  int i = blockIdx.x*blockDim.x+threadIdx.x; if (i >= B*D) return;
  int b = i/D, j = i%D;
  const float* gr = g + (long long)b*4*D;
  float cc = sigf(gr[D+j])*cd[i] + sigf(gr[j])*tanhf(gr[2*D+j]);
  float hh = sigf(gr[3*D+j])*tanhf(cc);
  cd[i] = cc;
  c1536[(long long)b*1536 + j] = (_Float16)hh;
}
// location conv: cum[B,2,T] (*) Wloc[A,2,31] pad15 -> pl[B*T,A]
__global__ void k_loc_conv(const float* __restrict__ cum, const float* __restrict__ w,
                           float* __restrict__ pl, int B, int T, int A){
  int i = blockIdx.x*blockDim.x+threadIdx.x; if (i >= B*T*A) return;
  int a = i % A; int bt = i / A; int t = bt % T; int b = bt / T;
  const float* c0 = cum + (long long)b*2*T;
  const float* c1 = c0 + T;
  const float* wr = w + (long long)a*62;
  float s = 0.f;
  #pragma unroll 4
  for (int k = 0; k < 31; ++k){
    int tt = t + k - 15;
    if (tt >= 0 && tt < T) s += c0[tt]*wr[k] + c1[tt]*wr[31+k];
  }
  pl[(long long)bt*A + a] = s;
}
// e[b,t] = sum_a v[a]*tanh(pq[b,a]+pm[bt,a]+pl[bt,a])
__global__ void k_energy(const float* __restrict__ pq, const float* __restrict__ pm,
                         const float* __restrict__ pl, const float* __restrict__ v,
                         float* __restrict__ e, int B, int T, int A){
  int i = blockIdx.x*blockDim.x+threadIdx.x; if (i >= B*T) return;
  int b = i / T;
  const float* pmr = pm + (long long)i*A;
  const float* plr = pl + (long long)i*A;
  const float* pqr = pq + (long long)b*A;
  float s = 0.f;
  for (int a = 0; a < A; ++a) s += v[a]*tanhf(pqr[a] + pmr[a] + plr[a]);
  e[i] = s;
}
// softmax over T per batch row; accumulate into both cum channels
__global__ void k_softmax(const float* __restrict__ e, float* __restrict__ al,
                          float* __restrict__ cum, int T){
  int b = blockIdx.x;
  __shared__ float red[256];
  float m = -1e30f;
  for (int t = threadIdx.x; t < T; t += blockDim.x) m = fmaxf(m, e[(long long)b*T + t]);
  red[threadIdx.x] = m; __syncthreads();
  for (int s = blockDim.x>>1; s > 0; s >>= 1){
    if (threadIdx.x < s) red[threadIdx.x] = fmaxf(red[threadIdx.x], red[threadIdx.x+s]);
    __syncthreads();
  }
  m = red[0]; __syncthreads();
  float sum = 0.f;
  for (int t = threadIdx.x; t < T; t += blockDim.x){
    float x = __expf(e[(long long)b*T + t] - m);
    al[(long long)b*T + t] = x; sum += x;
  }
  red[threadIdx.x] = sum; __syncthreads();
  for (int s = blockDim.x>>1; s > 0; s >>= 1){
    if (threadIdx.x < s) red[threadIdx.x] += red[threadIdx.x+s];
    __syncthreads();
  }
  float inv = 1.f / red[0];
  for (int t = threadIdx.x; t < T; t += blockDim.x){
    float a2 = al[(long long)b*T + t] * inv;
    al[(long long)b*T + t] = a2;
    cum[((long long)b*2    )*T + t] += a2;
    cum[((long long)b*2 + 1)*T + t] += a2;
  }
}
// ctx[b,e] = sum_t align[b,t]*mem[b,t,e]; scatter f16 into concat buffers
__global__ void k_ctx(const float* __restrict__ al, const float* __restrict__ mem,
                      float* __restrict__ ctx, _Float16* __restrict__ c768,
                      _Float16* __restrict__ c1536, int B, int T, int ENC){
  int i = blockIdx.x*blockDim.x+threadIdx.x; if (i >= B*ENC) return;
  int b = i/ENC, e2 = i%ENC;
  const float* a  = al + (long long)b*T;
  const float* mr = mem + (long long)b*T*ENC + e2;
  float s = 0.f;
  for (int t = 0; t < T; ++t) s += a[t]*mr[(long long)t*ENC];
  ctx[i] = s;
  c768 [(long long)b*768  + 256  + e2] = (_Float16)s;
  c1536[(long long)b*1536 + 1024 + e2] = (_Float16)s;
}
// teacher forcing: din[b,0:80] = mels[b,:,t] (pad to 96)
__global__ void k_set_decin(const float* __restrict__ mels, int t,
                            _Float16* __restrict__ din, int B, int NM, int T){
  int i = blockIdx.x*blockDim.x+threadIdx.x; if (i >= B*96) return;
  int b = i/96, c = i%96;
  din[i] = (_Float16)((c < NM) ? mels[((long long)b*NM + c)*T + t] : 0.f);
}
// mos[T,B,3,NM] -> mel_outputs[B,NM,3T]; gates from last mel index
__global__ void k_mos_to_mel(const float* __restrict__ mos, float* __restrict__ mel,
                             float* __restrict__ gates, int B, int T, int NM){
  int i = blockIdx.x*blockDim.x+threadIdx.x;
  if (i >= T*B*3*NM) return;
  int m = i % NM; int j = (i/NM) % 3; int b = (i/(3*NM)) % B; int t = i/(3*NM*B);
  float v = mos[i];
  mel[((long long)b*NM + m)*(3*T) + 3*t + j] = v;
  if (m == NM-1) gates[(long long)b*3*T + 3*t + j] = sigf(v);
}
// mel[B,NM,L] -> rows f16 [B*L, ldd] (pad NM->ldd)
__global__ void k_mel_rows(const float* __restrict__ mel, _Float16* __restrict__ rows,
                           int B, int NM, int L, int ldd){
  int i = blockIdx.x*blockDim.x+threadIdx.x; if (i >= B*L*ldd) return;
  int c = i % ldd; int r = i / ldd; int b = r / L; int l = r % L;
  rows[i] = (_Float16)((c < NM) ? mel[((long long)b*NM + c)*L + l] : 0.f);
}
// mel_postnet = mel + p2 (p2 is [B*L, NM] rows)
__global__ void k_post_final(const float* __restrict__ p2, const float* __restrict__ mel,
                             float* __restrict__ out, int B, int NM, int L){
  int i = blockIdx.x*blockDim.x+threadIdx.x; if (i >= B*NM*L) return;
  int l = i % L; int m = (i/L) % NM; int b = i/(NM*L);
  out[i] = mel[i] + p2[((long long)b*L + l)*NM + m];
}

// ---------------- host orchestration ----------------
#define NB(n) ((unsigned)(((n) + 255) / 256))

static inline void gemm(hipStream_t st, const _Float16* A, int lda,
                        const _Float16* W, int ldw, const float* bias,
                        float* C, int ldc, int M, int N, int K,
                        int tShift, int Tdim, int accum, int act,
                        _Float16* F = nullptr, int ldf = 0, int fOff = 0){
  int tilesN2 = ((N/16) + 1) / 2;
  int tiles = (M/16) * tilesN2;
  int blocks = (tiles + 7) / 8;
  k_wmma_gemm<<<blocks, 256, 0, st>>>(A, lda, W, ldw, bias, C, ldc,
                                      M, N, K, tShift, Tdim, accum, act,
                                      F, ldf, fOff);
}

extern "C" void kernel_launch(void* const* d_in, const int* in_sizes, int n_in,
                              void* d_out, int out_size, void* d_ws, size_t ws_size,
                              hipStream_t stream)
{
  (void)in_sizes; (void)n_in; (void)out_size; (void)ws_size;
  const int B=128, T=400, ENC=512, DEC=1024, NM=80, A=128;
  const int BT = B*T, L3 = 3*T, BL = B*L3;

  const int*   text  = (const int*)  d_in[0];
  const float* mels  = (const float*)d_in[1];
  const float* emb   = (const float*)d_in[2];
  const float* encw  = (const float*)d_in[3];
  const float* encb  = (const float*)d_in[4];
  const float* bng   = (const float*)d_in[5];
  const float* bnb   = (const float*)d_in[6];
  const float* Wih   = (const float*)d_in[7];
  const float* Whh   = (const float*)d_in[8];
  const float* lb    = (const float*)d_in[9];
  const float* preW1 = (const float*)d_in[10];
  const float* preb1 = (const float*)d_in[11];
  const float* preW2 = (const float*)d_in[12];
  const float* preb2 = (const float*)d_in[13];
  const float* attWq = (const float*)d_in[14];
  const float* attWm = (const float*)d_in[15];
  const float* attWl = (const float*)d_in[16];
  const float* attv  = (const float*)d_in[17];
  const float* decWi = (const float*)d_in[18];
  const float* decWh = (const float*)d_in[19];
  const float* decb  = (const float*)d_in[20];
  const float* outW  = (const float*)d_in[21];
  const float* outb  = (const float*)d_in[22];
  const float* pw1   = (const float*)d_in[23];
  const float* pb1   = (const float*)d_in[24];
  const float* pg    = (const float*)d_in[25];
  const float* pbta  = (const float*)d_in[26];
  const float* pw2   = (const float*)d_in[27];
  const float* pb2   = (const float*)d_in[28];

  float* melOut  = (float*)d_out;
  float* melPost = melOut + (size_t)B*NM*L3;
  float* gates   = melPost + (size_t)B*NM*L3;

  // bump allocator over workspace
  char* wsb = (char*)d_ws; size_t off = 0;
  auto alloc = [&](size_t bytes)->void*{
    off = (off + 255) & ~(size_t)255;
    void* p = wsb + off; off += bytes; return p;
  };
  // f16 weights
  _Float16* encwf  = (_Float16*)alloc((size_t)5*512*512*2);
  _Float16* wihf   = (_Float16*)alloc((size_t)6*1024*512*2);
  _Float16* whhf   = (_Float16*)alloc((size_t)6*1024*256*2);
  _Float16* preW1f = (_Float16*)alloc((size_t)256*96*2);
  _Float16* preW2f = (_Float16*)alloc((size_t)256*256*2);
  _Float16* attWqf = (_Float16*)alloc((size_t)128*1024*2);
  _Float16* attWmf = (_Float16*)alloc((size_t)128*512*2);
  _Float16* decWif = (_Float16*)alloc((size_t)4096*768*2);
  _Float16* decWhf = (_Float16*)alloc((size_t)4096*1024*2);
  _Float16* outWf  = (_Float16*)alloc((size_t)240*1536*2);
  _Float16* pw1f   = (_Float16*)alloc((size_t)5*512*96*2);
  _Float16* pw2f   = (_Float16*)alloc((size_t)5*80*512*2);
  // activations
  _Float16* xemb   = (_Float16*)alloc((size_t)BT*512*2);
  float*    convO  = (float*)   alloc((size_t)BT*512*4);
  float*    mu1    = (float*)   alloc(512*4);
  float*    inv1   = (float*)   alloc(512*4);
  _Float16* lbufA  = (_Float16*)alloc((size_t)BT*512*2);
  _Float16* lbufB  = (_Float16*)alloc((size_t)BT*512*2);
  float*    G      = (float*)   alloc((size_t)BT*1024*4);
  float*    cb     = (float*)   alloc((size_t)B*256*4);
  _Float16* hf16   = (_Float16*)alloc((size_t)B*256*2);
  float*    mem    = (float*)   alloc((size_t)BT*512*4);
  _Float16* memh   = (_Float16*)alloc((size_t)BT*512*2);
  float*    pmB    = (float*)   alloc((size_t)BT*128*4);
  float*    plB    = (float*)   alloc((size_t)BT*128*4);
  float*    ebuf   = (float*)   alloc((size_t)BT*4);
  float*    align  = (float*)   alloc((size_t)BT*4);
  float*    cum    = (float*)   alloc((size_t)B*2*T*4);
  float*    pq     = (float*)   alloc((size_t)B*128*4);
  float*    pr1    = (float*)   alloc((size_t)B*256*4);
  _Float16* pr1f   = (_Float16*)alloc((size_t)B*256*2);
  float*    pr2    = (float*)   alloc((size_t)B*256*4);
  _Float16* c768   = (_Float16*)alloc((size_t)B*768*2);
  _Float16* c1536  = (_Float16*)alloc((size_t)B*1536*2);
  float*    gdec   = (float*)   alloc((size_t)B*4096*4);
  float*    cd     = (float*)   alloc((size_t)B*1024*4);
  float*    ctx    = (float*)   alloc((size_t)B*512*4);
  _Float16* din    = (_Float16*)alloc((size_t)B*96*2);
  float*    mos    = (float*)   alloc((size_t)T*B*240*4);
  _Float16* melTr  = (_Float16*)alloc((size_t)BL*96*2);
  float*    pbuf   = (float*)   alloc((size_t)BL*512*4);
  float*    mu2    = (float*)   alloc(512*4);
  float*    inv2   = (float*)   alloc(512*4);
  _Float16* pT     = (_Float16*)alloc((size_t)BL*512*2);
  float*    p2     = (float*)   alloc((size_t)BL*80*4);

  // ---- weight f16 prep (deterministic each call) ----
  for (int k = 0; k < 5; ++k)
    k_convw_slice<<<NB(512*512),256,0,stream>>>(encw, encwf + (size_t)k*512*512, 512, 512, 5, k, 512);
  k_cvt_pad<<<NB(6144*512),256,0,stream>>>(Wih,  512, wihf, 512, 6144, 512);
  k_cvt_pad<<<NB(6144*256),256,0,stream>>>(Whh,  256, whhf, 256, 6144, 256);
  k_cvt_pad<<<NB(256*96),  256,0,stream>>>(preW1, 80, preW1f, 96, 256, 80);
  k_cvt_pad<<<NB(256*256), 256,0,stream>>>(preW2,256, preW2f,256, 256,256);
  k_cvt_pad<<<NB(128*1024),256,0,stream>>>(attWq,1024,attWqf,1024,128,1024);
  k_cvt_pad<<<NB(128*512), 256,0,stream>>>(attWm, 512,attWmf, 512,128, 512);
  k_cvt_pad<<<NB(4096*768),256,0,stream>>>(decWi, 768,decWif, 768,4096,768);
  k_cvt_pad<<<NB(4096*1024),256,0,stream>>>(decWh,1024,decWhf,1024,4096,1024);
  k_cvt_pad<<<NB(240*1536),256,0,stream>>>(outW, 1536,outWf, 1536,240,1536);
  for (int k = 0; k < 5; ++k)
    k_convw_slice<<<NB(512*96),256,0,stream>>>(pw1, pw1f + (size_t)k*512*96, 512, 80, 5, k, 96);
  for (int k = 0; k < 5; ++k)
    k_convw_slice<<<NB(80*512),256,0,stream>>>(pw2, pw2f + (size_t)k*80*512, 80, 512, 5, k, 512);

  // ---- encoder: embedding -> conv5 (5 shifted GEMMs) -> BN ----
  k_embed<<<NB(BT*512),256,0,stream>>>(text, emb, xemb, BT, 512);
  for (int k = 0; k < 5; ++k)
    gemm(stream, xemb, 512, encwf + (size_t)k*512*512, 512,
         (k==0)?encb:nullptr, convO, 512, BT, 512, 512,
         k-2, T, (k>0)?1:0, (k==4)?1:0);
  k_bn_stats<<<512,256,0,stream>>>(convO, BT, 512, mu1, inv1, 1e-5f);
  k_enc_bn<<<NB(BT*512),256,0,stream>>>(convO, mu1, inv1, bng, bnb, lbufA, B, T, 512);

  // ---- 3-layer biLSTM ----
  _Float16* lin = lbufA; _Float16* lout = lbufB;
  for (int l = 0; l < 3; ++l){
    for (int d = 0; d < 2; ++d){
      int ld6 = l*2 + d;
      k_fill32<<<NB(B*256),256,0,stream>>>(cb, 0.f, B*256);
      k_fill16<<<NB(B*256),256,0,stream>>>(hf16, B*256);
      // precompute x@Wih^T + b for all timesteps
      gemm(stream, lin, 512, wihf + (size_t)ld6*1024*512, 512,
           lb + (size_t)ld6*1024, G, 1024, BT, 1024, 512, 0, BT, 0, 0);
      for (int s = 0; s < T; ++s){
        int t = d ? (T-1-s) : s;
        gemm(stream, hf16, 256, whhf + (size_t)ld6*1024*256, 256, nullptr,
             G + (size_t)t*B*1024, 1024, B, 1024, 256, 0, B, 1, 0);
        k_lstm_point<<<NB(B*256),256,0,stream>>>(
            G + (size_t)t*B*1024, cb, hf16, lout,
            (l==2)?mem:nullptr, (l==2)?memh:nullptr, t, B, 256, d*256, T);
      }
    }
    _Float16* tmp = lin; lin = lout; lout = tmp;
  }

  // pm = memory @ att_Wm^T (constant over decoder steps)
  gemm(stream, memh, 512, attWmf, 512, nullptr, pmB, 128, BT, 128, 512, 0, BT, 0, 0);

  // ---- decoder init ----
  k_fill32<<<NB(B*2*T),256,0,stream>>>(cum, 0.f, B*2*T);
  k_fill32<<<NB(B*1024),256,0,stream>>>(cd, 0.f, B*1024);
  k_fill16<<<NB(B*1536),256,0,stream>>>(c1536, B*1536);
  k_fill16<<<NB(B*768),256,0,stream>>>(c768, B*768);
  k_fill16<<<NB(B*96),256,0,stream>>>(din, B*96);

  // ---- decoder scan (teacher forcing) ----
  for (int t = 0; t < T; ++t){
    // prenet (f16 outputs fused into GEMM epilogue)
    gemm(stream, din, 96, preW1f, 96, preb1, pr1, 256, B, 256, 96, 0, B, 0, 1,
         pr1f, 256, 0);
    gemm(stream, pr1f, 256, preW2f, 256, preb2, pr2, 256, B, 256, 256, 0, B, 0, 1,
         c768, 768, 0);
    // location-sensitive attention
    k_loc_conv<<<NB(B*T*128),256,0,stream>>>(cum, attWl, plB, B, T, A);
    gemm(stream, c1536, 1536, attWqf, 1024, nullptr, pq, 128, B, 128, 1024, 0, B, 0, 0);
    k_energy<<<NB(BT),256,0,stream>>>(pq, pmB, plB, attv, ebuf, B, T, A);
    k_softmax<<<B,256,0,stream>>>(ebuf, align, cum, T);
    k_ctx<<<NB(B*512),256,0,stream>>>(align, mem, ctx, c768, c1536, B, T, ENC);
    // decoder LSTM cell
    gemm(stream, c768, 768, decWif, 768, decb, gdec, 4096, B, 4096, 768, 0, B, 0, 0);
    gemm(stream, c1536, 1536, decWhf, 1024, nullptr, gdec, 4096, B, 4096, 1024, 0, B, 1, 0);
    k_dec_point<<<NB(B*1024),256,0,stream>>>(gdec, cd, c1536, B, DEC);
    // output projection
    gemm(stream, c1536, 1536, outWf, 1536, outb, mos + (size_t)t*B*240, 240,
         B, 240, 1536, 0, B, 0, 0);
    // next-step teacher-forcing input
    k_set_decin<<<NB(B*96),256,0,stream>>>(mels, t, din, B, NM, T);
  }
  k_mos_to_mel<<<NB(T*B*3*NM),256,0,stream>>>(mos, melOut, gates, B, T, NM);

  // ---- postnet ----
  k_mel_rows<<<NB(BL*96),256,0,stream>>>(melOut, melTr, B, NM, L3, 96);
  for (int k = 0; k < 5; ++k)
    gemm(stream, melTr, 96, pw1f + (size_t)k*512*96, 96, (k==0)?pb1:nullptr,
         pbuf, 512, BL, 512, 96, k-2, L3, (k>0)?1:0, 0);
  k_bn_stats<<<512,256,0,stream>>>(pbuf, BL, 512, mu2, inv2, 1e-5f);
  k_bn_tanh<<<NB(BL*512),256,0,stream>>>(pbuf, mu2, inv2, pg, pbta, pT, BL, 512);
  for (int k = 0; k < 5; ++k)
    gemm(stream, pT, 512, pw2f + (size_t)k*80*512, 512, (k==0)?pb2:nullptr,
         p2, 80, BL, 80, 512, k-2, L3, (k>0)?1:0, 0);
  k_post_final<<<NB(B*NM*L3),256,0,stream>>>(p2, melOut, melPost, B, NM, L3);
}
